// GPTBlockBase_7713761264293
// MI455X (gfx1250) — matrix operand order repaired and checked by reading, not compiled, not run on hardware
//
#include <hip/hip_runtime.h>
#include <math.h>

// GPT block: x -> causal MHA -> +res, LN1 -> GELU FFN -> +res, LN2
// B=2, T=2048, D=1024, H=16, DK=64, DFF=4096. All fp32 in/out.
// f16 WMMA (f32 accum) for all matmuls; flash-attention streaming softmax;
// V stored transposed so PV fragments are contiguous b128 loads.

typedef _Float16 h16;
typedef __attribute__((ext_vector_type(16))) _Float16 v16h;
typedef __attribute__((ext_vector_type(8)))  float    v8f;

#define B_   2
#define T_   2048
#define D_   1024
#define H_   16
#define DK_  64
#define DFF_ 4096
#define M_   (B_ * T_)

#define BM 128
#define BN 128
#define BK 32
#define LDA 40   // padded LDS row stride (halfs), multiple of 8 -> 16B aligned rows
#define LDB 40

__device__ inline v8f wmma_f16(v16h a, v16h b, v8f c) {
  // D = A(16x32 f16) * B(32x16 f16) + C(16x16 f32)
  return __builtin_amdgcn_wmma_f32_16x16x32_f16(
      false, a, false, b, (short)0, c, false, false);
}

// A-matrix fragment (16x32, contiguous K per 16-lane group):
// lane group g: halfs j -> k=16g+j
__device__ inline v16h frag_a_lds(const h16* base, int row, int ld, int g) {
  union { v16h v; uint4 q[2]; } u;
  const h16* p = base + row * ld + 16 * g;
  u.q[0] = *(const uint4*)(p);
  u.q[1] = *(const uint4*)(p + 8);
  return u.v;
}

// B-matrix fragment (32x16) read from a transposed tile Bt[n][k]:
// lane group g holds contiguous k = 16g .. 16g+15
__device__ inline v16h frag_b_lds(const h16* base, int row, int ld, int g) {
  union { v16h v; uint4 q[2]; } u;
  const h16* p = base + row * ld + 16 * g;
  u.q[0] = *(const uint4*)(p);
  u.q[1] = *(const uint4*)(p + 8);
  return u.v;
}

// ---------------------------------------------------------------------------
// GEMM: C = act(A[M,K] * W[K,N] + bias), fp32 inputs converted to f16 in LDS.
// Block tile 128x128, 8 waves each computing 32x64 (2x4 WMMA tiles).
// MODE 0: fp32 row-major out
// MODE 1: fp32 row-major out + exact-erf GELU
// MODE 2: f16 out scattered to [B,H,T,DK]   (Q/K)
// MODE 3: f16 out scattered to [B,H,DK,T]   (V transposed)
// ---------------------------------------------------------------------------
template <int MODE>
__global__ __launch_bounds__(256) void gemm_wmma(
    const float* __restrict__ A, const float* __restrict__ W,
    const float* __restrict__ bias, void* __restrict__ Cout,
    int M, int N, int K, int Tdim) {
  __shared__ __align__(16) h16 sA[BM * LDA];
  __shared__ __align__(16) h16 sB[BN * LDB];

  const int tid  = threadIdx.x;
  const int wave = tid >> 5;
  const int lane = tid & 31;
  const int g    = lane >> 4;
  const int ln   = lane & 15;
  const int m0   = blockIdx.y * BM;
  const int n0   = blockIdx.x * BN;
  const int wm   = (wave >> 1) * 32;  // 4 waves along M
  const int wn   = (wave & 1) * 64;   // 2 waves along N

  v8f zero = {0.f, 0.f, 0.f, 0.f, 0.f, 0.f, 0.f, 0.f};
  v8f acc[2][4];
  for (int i = 0; i < 2; ++i)
    for (int j = 0; j < 4; ++j) acc[i][j] = zero;

  for (int k0 = 0; k0 < K; k0 += BK) {
    // Stage A tile (BM x BK) as f16, row-major
    for (int f4 = tid; f4 < (BM * BK) / 4; f4 += 256) {
      int r  = f4 >> 3;            // BK/4 = 8 float4 per row
      int c4 = (f4 & 7) << 2;
      float4 v = *(const float4*)(A + (size_t)(m0 + r) * K + k0 + c4);
      union { h16 h[4]; uint2 u; } t;
      t.h[0] = (h16)v.x; t.h[1] = (h16)v.y; t.h[2] = (h16)v.z; t.h[3] = (h16)v.w;
      *(uint2*)&sA[r * LDA + c4] = t.u;
    }
    // Stage B tile (BK x BN) as f16, TRANSPOSED: sB[n][k]
    for (int f4 = tid; f4 < (BK * BN) / 4; f4 += 256) {
      int kr = f4 >> 5;            // BN/4 = 32 float4 per row
      int c4 = (f4 & 31) << 2;
      float4 v = *(const float4*)(W + (size_t)(k0 + kr) * N + n0 + c4);
      sB[(c4 + 0) * LDB + kr] = (h16)v.x;
      sB[(c4 + 1) * LDB + kr] = (h16)v.y;
      sB[(c4 + 2) * LDB + kr] = (h16)v.z;
      sB[(c4 + 3) * LDB + kr] = (h16)v.w;
    }
    __syncthreads();

    v16h af[2], bf[4];
    af[0] = frag_a_lds(sA, wm + ln,      LDA, g);
    af[1] = frag_a_lds(sA, wm + 16 + ln, LDA, g);
    for (int tn = 0; tn < 4; ++tn)
      bf[tn] = frag_b_lds(sB, wn + tn * 16 + ln, LDB, g);
    for (int tm = 0; tm < 2; ++tm)
      for (int tn = 0; tn < 4; ++tn)
        acc[tm][tn] = wmma_f16(af[tm], bf[tn], acc[tm][tn]);
    __syncthreads();
  }

  // Epilogue. C layout: VGPR r, lane: row_local = 8g + r, col_local = ln.
  for (int tm = 0; tm < 2; ++tm)
    for (int tn = 0; tn < 4; ++tn)
      for (int r = 0; r < 8; ++r) {
        int row = m0 + wm + tm * 16 + 8 * g + r;
        int col = n0 + wn + tn * 16 + ln;
        float val = acc[tm][tn][r] + bias[col];
        if (MODE == 1)
          val = 0.5f * val * (1.0f + erff(val * 0.70710678118654752f));
        if (MODE == 0 || MODE == 1) {
          ((float*)Cout)[(size_t)row * N + col] = val;
        } else {
          int bi = row / Tdim, t = row - bi * Tdim;
          int hh = col >> 6, dk = col & 63;
          size_t idx = (MODE == 2)
              ? ((((size_t)bi * H_ + hh) * Tdim + t) * DK_ + dk)     // [B,H,T,DK]
              : ((((size_t)bi * H_ + hh) * DK_ + dk) * Tdim + t);    // [B,H,DK,T]
          ((h16*)Cout)[idx] = (h16)val;
        }
      }
}

// ---------------------------------------------------------------------------
// Flash attention, causal. One wave = one 16-row query tile; 32-key steps.
// Q,K: f16 [B,H,T,DK]; Vt: f16 [B,H,DK,T]; out: fp32 [B,T,D] (heads merged).
// ---------------------------------------------------------------------------
__global__ __launch_bounds__(256) void attn_kernel(
    const h16* __restrict__ Qh, const h16* __restrict__ Kh,
    const h16* __restrict__ Vt, const unsigned char* __restrict__ pad,
    float* __restrict__ out) {
  __shared__ __align__(16) h16 sP[8 * 16 * 40];  // per-wave 16x32 P tile, padded

  const int wave = threadIdx.x >> 5;
  const int lane = threadIdx.x & 31;
  const int g    = lane >> 4;
  const int ln   = lane & 15;
  const int bh   = blockIdx.x >> 4;            // 16 blocks per (b,h)
  const int bi   = bh >> 4;                    // H = 16
  const int hd   = bh & 15;
  const int qtile = (blockIdx.x & 15) * 8 + wave;
  const int q0    = qtile * 16;

  const h16* Qb = Qh + (size_t)bh * T_ * DK_;
  const h16* Kb = Kh + (size_t)bh * T_ * DK_;
  const h16* Vb = Vt + (size_t)bh * DK_ * T_;
  const unsigned char* padb = pad + (size_t)bi * T_;
  h16* sPw = sP + wave * 640;

  // Q fragments (A layout), d split 0..31 / 32..63
  v16h qf[2];
  {
    const h16* qrow = Qb + (size_t)(q0 + ln) * DK_;
    for (int f = 0; f < 2; ++f) {
      union { v16h v; uint4 q[2]; } u;
      u.q[0] = *(const uint4*)(qrow + 32 * f + 16 * g);
      u.q[1] = *(const uint4*)(qrow + 32 * f + 16 * g + 8);
      qf[f] = u.v;
    }
  }

  v8f zero = {0.f, 0.f, 0.f, 0.f, 0.f, 0.f, 0.f, 0.f};
  v8f o[4]; o[0] = zero; o[1] = zero; o[2] = zero; o[3] = zero;
  float mrun[8], lrun[8];
  for (int r = 0; r < 8; ++r) { mrun[r] = -1e30f; lrun[r] = 0.f; }

  const int kend = q0 + 16;
  for (int kb = 0; kb < kend; kb += 32) {
    // S = Q * K^T for 32 keys (two 16x16 C tiles)
    v8f s0 = zero, s1 = zero;
    for (int f = 0; f < 2; ++f) {
      union { v16h v; uint4 q[2]; } u0, u1;
      const h16* kr0 = Kb + (size_t)(kb + ln) * DK_ + 32 * f + 16 * g;
      const h16* kr1 = Kb + (size_t)(kb + 16 + ln) * DK_ + 32 * f + 16 * g;
      u0.q[0] = *(const uint4*)(kr0); u0.q[1] = *(const uint4*)(kr0 + 8);
      u1.q[0] = *(const uint4*)(kr1); u1.q[1] = *(const uint4*)(kr1 + 8);
      s0 = wmma_f16(qf[f], u0.v, s0);
      s1 = wmma_f16(qf[f], u1.v, s1);
    }
    // online softmax update per row (row = q0 + 8g + r, col = lane group)
    for (int r = 0; r < 8; ++r) {
      int row = q0 + 8 * g + r;
      int k0i = kb + ln, k1i = kb + 16 + ln;
      float v0 = s0[r] * 0.125f;  // 1/sqrt(64)
      float v1 = s1[r] * 0.125f;
      if (k0i > row || !padb[k0i]) v0 = -1e30f;
      if (k1i > row || !padb[k1i]) v1 = -1e30f;
      float mx = fmaxf(v0, v1);
      mx = fmaxf(mx, __shfl_xor(mx, 1, 32));
      mx = fmaxf(mx, __shfl_xor(mx, 2, 32));
      mx = fmaxf(mx, __shfl_xor(mx, 4, 32));
      mx = fmaxf(mx, __shfl_xor(mx, 8, 32));
      float mnew = fmaxf(mrun[r], mx);
      float alpha = expf(mrun[r] - mnew);
      float p0 = expf(v0 - mnew);
      float p1 = expf(v1 - mnew);
      float ps = p0 + p1;
      ps += __shfl_xor(ps, 1, 32);
      ps += __shfl_xor(ps, 2, 32);
      ps += __shfl_xor(ps, 4, 32);
      ps += __shfl_xor(ps, 8, 32);
      lrun[r] = lrun[r] * alpha + ps;
      mrun[r] = mnew;
      o[0][r] *= alpha; o[1][r] *= alpha; o[2][r] *= alpha; o[3][r] *= alpha;
      // write P (f16) to wave-private LDS, row-major [16][32] padded 40
      sPw[(8 * g + r) * 40 + ln]      = (h16)p0;
      sPw[(8 * g + r) * 40 + 16 + ln] = (h16)p1;
    }
    // O += P(16x32) * V(32x16 per d-tile); P as A-frag from LDS, V contiguous
    v16h pf = frag_a_lds(sPw, ln, 40, g);
    for (int dt = 0; dt < 4; ++dt) {
      const h16* vrow = Vb + (size_t)(dt * 16 + ln) * T_ + kb + 16 * g;
      union { v16h v; uint4 q[2]; } u;
      u.q[0] = *(const uint4*)(vrow);
      u.q[1] = *(const uint4*)(vrow + 8);
      o[dt] = wmma_f16(pf, u.v, o[dt]);
    }
  }

  // normalize and store merged-head fp32 [B,T,D]
  for (int r = 0; r < 8; ++r) {
    float inv = 1.0f / fmaxf(lrun[r], 1e-30f);
    int t = q0 + 8 * g + r;
    float* orow = out + ((size_t)bi * T_ + t) * D_ + hd * DK_;
    for (int dt = 0; dt < 4; ++dt)
      orow[dt * 16 + ln] = o[dt][r] * inv;
  }
}

// ---------------------------------------------------------------------------
// LayerNorm with fused residual: out = gamma * norm(a + res) + beta
// ---------------------------------------------------------------------------
__global__ __launch_bounds__(256) void ln_kernel(
    const float* __restrict__ a, const float* __restrict__ res,
    const float* __restrict__ gamma, const float* __restrict__ beta,
    float* __restrict__ out, int Dd) {
  const int row = blockIdx.x;
  const float* pa = a + (size_t)row * Dd;
  const float* pr = res + (size_t)row * Dd;
  float s1 = 0.f, s2 = 0.f;
  for (int i = threadIdx.x; i < Dd; i += 256) {
    float v = pa[i] + pr[i];
    s1 += v; s2 += v * v;
  }
  for (int off = 16; off > 0; off >>= 1) {
    s1 += __shfl_xor(s1, off, 32);
    s2 += __shfl_xor(s2, off, 32);
  }
  __shared__ float r1[8], r2[8];
  const int wave = threadIdx.x >> 5;
  if ((threadIdx.x & 31) == 0) { r1[wave] = s1; r2[wave] = s2; }
  __syncthreads();
  if (threadIdx.x == 0) {
    float t1 = 0.f, t2 = 0.f;
    for (int i = 0; i < 8; ++i) { t1 += r1[i]; t2 += r2[i]; }
    r1[0] = t1; r2[0] = t2;
  }
  __syncthreads();
  float mean = r1[0] / (float)Dd;
  float var  = r2[0] / (float)Dd - mean * mean;
  float rstd = rsqrtf(var + 1e-5f);
  for (int i = threadIdx.x; i < Dd; i += 256) {
    float v = pa[i] + pr[i];
    out[(size_t)row * Dd + i] = gamma[i] * (v - mean) * rstd + beta[i];
  }
}

// ---------------------------------------------------------------------------
extern "C" void kernel_launch(void* const* d_in, const int* in_sizes, int n_in,
                              void* d_out, int out_size, void* d_ws,
                              size_t ws_size, hipStream_t stream) {
  const float* x  = (const float*)d_in[0];
  const unsigned char* mask = (const unsigned char*)d_in[1];
  const float* wq = (const float*)d_in[2];
  const float* bq = (const float*)d_in[3];
  const float* wk = (const float*)d_in[4];
  const float* bk = (const float*)d_in[5];
  const float* wv = (const float*)d_in[6];
  const float* bv = (const float*)d_in[7];
  const float* wo = (const float*)d_in[8];
  const float* bo = (const float*)d_in[9];
  const float* w1 = (const float*)d_in[10];
  const float* b1 = (const float*)d_in[11];
  const float* w2 = (const float*)d_in[12];
  const float* b2 = (const float*)d_in[13];
  const float* g1 = (const float*)d_in[14];
  const float* be1 = (const float*)d_in[15];
  const float* g2 = (const float*)d_in[16];
  const float* be2 = (const float*)d_in[17];

  char* ws = (char*)d_ws;
  const size_t MB = 1u << 20;
  h16*   Qh    = (h16*)(ws);                 // 8 MB  [B,H,T,DK] f16
  h16*   Kh    = (h16*)(ws + 8 * MB);        // 8 MB
  h16*   Vt    = (h16*)(ws + 16 * MB);       // 8 MB  [B,H,DK,T] f16
  float* attnO = (float*)(ws + 24 * MB);     // 16 MB [B,T,D]
  float* attnP = (float*)(ws + 40 * MB);     // 16 MB
  float* hbuf  = (float*)(ws);               // 16 MB (reuses Q/K region)
  float* ffn1  = (float*)(ws + 16 * MB);     // 64 MB (reuses V/attnO/attnP)
  float* ffn2  = (float*)(ws + 80 * MB);     // 16 MB -> peak 96 MB

  dim3 blk(256);
  dim3 gQKV(D_ / BN, M_ / BM);     // (8, 32)
  dim3 gFFN1(DFF_ / BN, M_ / BM);  // (32, 32)

  // 1-3: QKV projections (f16 out, head layouts; V pre-transposed)
  gemm_wmma<2><<<gQKV, blk, 0, stream>>>(x, wq, bq, Qh, M_, D_, D_, T_);
  gemm_wmma<2><<<gQKV, blk, 0, stream>>>(x, wk, bk, Kh, M_, D_, D_, T_);
  gemm_wmma<3><<<gQKV, blk, 0, stream>>>(x, wv, bv, Vt, M_, D_, D_, T_);
  // 4: causal flash attention -> fp32 [B,T,D]
  attn_kernel<<<dim3(B_ * H_ * 16), blk, 0, stream>>>(Qh, Kh, Vt, mask, attnO);
  // 5: output projection
  gemm_wmma<0><<<gQKV, blk, 0, stream>>>(attnO, wo, bo, attnP, M_, D_, D_, T_);
  // 6: h = LN(x + attn)
  ln_kernel<<<dim3(M_), blk, 0, stream>>>(x, attnP, g1, be1, hbuf, D_);
  // 7: ffn1 = GELU(h @ w1 + b1)
  gemm_wmma<1><<<gFFN1, blk, 0, stream>>>(hbuf, w1, b1, ffn1, M_, DFF_, D_, T_);
  // 8: ffn2 = ffn1 @ w2 + b2
  gemm_wmma<0><<<gQKV, blk, 0, stream>>>(ffn1, w2, b2, ffn2, M_, D_, DFF_, T_);
  // 9: out = LN(h + ffn2)
  ln_kernel<<<dim3(M_), blk, 0, stream>>>(hbuf, ffn2, g2, be2, (float*)d_out, D_);
}
